// AttnResBlock_57698590655209
// MI455X (gfx1250) — matrix operand order repaired and checked by reading, not compiled, not run on hardware
//
#include <hip/hip_runtime.h>
#include <hip/hip_bf16.h>
#include <stdint.h>

// Problem constants (from reference): B=64, CH1=3, CH2=4, HID=16, T=16384, E=512
#define B_   64
#define T_   16384
#define E_   512

typedef unsigned short u16;
typedef __attribute__((ext_vector_type(16))) __bf16 v16bf;
typedef __attribute__((ext_vector_type(8)))  float  v8f;
typedef __attribute__((ext_vector_type(4)))  int    v4i;

// ---------------------------------------------------------------------------
// CDNA5 async global->LDS copy (probe-verified prototype:
//   (v4i addrspace(1)* src, v4i addrspace(3)* dst, imm offset, imm cpol))
// ---------------------------------------------------------------------------
#if defined(__HIP_DEVICE_COMPILE__) && defined(__gfx1250__) && \
    __has_builtin(__builtin_amdgcn_global_load_async_to_lds_b128)
#define USE_ASYNC 1
#else
#define USE_ASYNC 0
#endif

#if USE_ASYNC
typedef __attribute__((address_space(1))) v4i* gp_v4i;
typedef __attribute__((address_space(3))) v4i* lp_v4i;
#endif

__device__ __forceinline__ void cp16(u16* dst_lds, const u16* src_g) {
#if USE_ASYNC
  __builtin_amdgcn_global_load_async_to_lds_b128((gp_v4i)src_g, (lp_v4i)dst_lds,
                                                 0, 0);
#else
  *(uint4*)dst_lds = *(const uint4*)src_g;
#endif
}

__device__ __forceinline__ void wait_async() {
#if USE_ASYNC
#if __has_builtin(__builtin_amdgcn_s_wait_asynccnt)
  __builtin_amdgcn_s_wait_asynccnt(0);
#else
  asm volatile("s_wait_asynccnt 0" ::: "memory");
#endif
#endif
}

// float -> bf16 round-to-nearest-even
__device__ __forceinline__ u16 f2bf(float f) {
  uint32_t u = __float_as_uint(f);
  u += 0x7fffu + ((u >> 16) & 1u);
  return (u16)(u >> 16);
}

__device__ __forceinline__ v8f wmma_bf16(v16bf a, v16bf b, v8f c) {
  // 8 args: (neg_a, A, neg_b, B, c_mod, C, reuse_a, reuse_b)
  return __builtin_amdgcn_wmma_f32_16x16x32_bf16(false, a, false, b,
                                                 (short)0, c, false, false);
}

// A fragment (16x32 bf16, M x K): lane m<16 holds k=[0..7],[16..23]; lane m+16
// holds k=[8..15],[24..31].  Row-major LDS tile, padded stride 72 halves.
__device__ __forceinline__ v16bf ldfragA(const u16* tile, int row, int kk, int laneHi) {
  v16bf r;
  const u16* p = tile + row * 72 + kk + laneHi * 8;
  ((uint4*)&r)[0] = *(const uint4*)p;
  ((uint4*)&r)[1] = *(const uint4*)(p + 16);
  return r;
}

// B fragment (32x16 bf16, K x N) from an N-major ("B^T row-major", K-contig)
// tile: lane n<16 holds k=[0..7],[16..23]; lane n+16 holds k=[8..15],[24..31].
__device__ __forceinline__ v16bf ldfragB(const u16* tile, int row, int stride,
                                         int kk, int laneHi) {
  v16bf r;
  const u16* p = tile + row * stride + kk + laneHi * 8;
  ((uint4*)&r)[0] = *(const uint4*)p;
  ((uint4*)&r)[1] = *(const uint4*)(p + 16);
  return r;
}

// ---------------------------------------------------------------------------
// Kernel 1: q = W1 @ p, cc = W2 @ c, emitted as bf16 rows (b*16+o, t)
// ---------------------------------------------------------------------------
__global__ __launch_bounds__(256) void k_qcc(const float* __restrict__ x,
                                             const float* __restrict__ W1,
                                             const float* __restrict__ W2,
                                             u16* __restrict__ qbf,
                                             u16* __restrict__ ccbf) {
  int idx = blockIdx.x * 256 + threadIdx.x;   // over B*T
  int b = idx >> 14, t = idx & (T_ - 1);
  const float* xb = x + (size_t)b * 7 * T_ + t;
  float p0 = xb[0],      p1 = xb[T_],     p2 = xb[2 * T_];
  float c0 = xb[3 * T_], c1 = xb[4 * T_], c2 = xb[5 * T_], c3 = xb[6 * T_];
  size_t base = ((size_t)b * 16) * T_ + t;
#pragma unroll
  for (int o = 0; o < 16; ++o) {
    float qv = W1[o * 3 + 0] * p0 + W1[o * 3 + 1] * p1 + W1[o * 3 + 2] * p2;
    float cv = W2[o * 4 + 0] * c0 + W2[o * 4 + 1] * c1 +
               W2[o * 4 + 2] * c2 + W2[o * 4 + 3] * c3;
    qbf[base + (size_t)o * T_]  = f2bf(qv);
    ccbf[base + (size_t)o * T_] = f2bf(cv);
  }
}

// ---------------------------------------------------------------------------
// Kernel 2: f32 -> bf16 weight conversion (4 elems / thread)
// ---------------------------------------------------------------------------
__global__ __launch_bounds__(256) void k_cvt4(const float* __restrict__ in,
                                              u16* __restrict__ out, int n4) {
  int i = blockIdx.x * 256 + threadIdx.x;
  if (i >= n4) return;
  float4 v = ((const float4*)in)[i];
  union { u16 h[4]; uint2 u; } r;
  r.h[0] = f2bf(v.x); r.h[1] = f2bf(v.y); r.h[2] = f2bf(v.z); r.h[3] = f2bf(v.w);
  ((uint2*)out)[i] = r.u;
}

// ---------------------------------------------------------------------------
// Kernel 3: NT GEMM  C[M,N] = A[M,K] * B[N,K]^T + bias[N]  (bf16 in, f32 out)
// Block tile 64(M) x 128(N), kc=64, double-buffered async LDS.
// 8 waves, each computing 32x32 (4 accumulators, A/B fragments reused 2x
// => 2 ds_load_b128 per WMMA instead of 3).
// ---------------------------------------------------------------------------
__global__ __launch_bounds__(256) void k_gemm_nt(const u16* __restrict__ A,
                                                 const u16* __restrict__ Bw,
                                                 const float* __restrict__ bias,
                                                 float* __restrict__ C,
                                                 int M, int N, int K) {
  __shared__ __align__(16) u16 sA[2][64 * 72];
  __shared__ __align__(16) u16 sB[2][128 * 72];
  const int tid = threadIdx.x;
  const int lane = tid & 31, w = tid >> 5;
  const int laneHi = lane >> 4, ln = lane & 15;
  const int gm0 = blockIdx.y * 64, gn0 = blockIdx.x * 128;

  auto issue = [&](int buf, int k0) {
#pragma unroll
    for (int i = 0; i < 2; ++i) {       // A: 64 rows x 8 x 16B segments
      int s = tid + i * 256;
      int r = s >> 3, sg = s & 7;
      cp16(&sA[buf][r * 72 + sg * 8], A + (size_t)(gm0 + r) * K + k0 + sg * 8);
    }
#pragma unroll
    for (int i = 0; i < 4; ++i) {       // B: 128 rows x 8 x 16B segments
      int s = tid + i * 256;
      int r = s >> 3, sg = s & 7;
      cp16(&sB[buf][r * 72 + sg * 8], Bw + (size_t)(gn0 + r) * K + k0 + sg * 8);
    }
  };

  v8f acc00 = {}, acc01 = {}, acc10 = {}, acc11 = {};
  issue(0, 0);
  const int nk  = K >> 6;
  const int rA0 = (w & 1) * 32 + ln, rA1 = rA0 + 16;
  const int rB0 = (w >> 1) * 32 + ln, rB1 = rB0 + 16;
  for (int kt = 0; kt < nk; ++kt) {
    const int buf = kt & 1;
    wait_async();
    __syncthreads();
    if (kt + 1 < nk) issue(buf ^ 1, (kt + 1) << 6);
    const u16* Ab = sA[buf];
    const u16* Bb = sB[buf];
#pragma unroll
    for (int kk = 0; kk < 64; kk += 32) {
      v16bf b0 = ldfragB(Bb, rB0, 72, kk, laneHi);
      v16bf b1 = ldfragB(Bb, rB1, 72, kk, laneHi);
      v16bf a0 = ldfragA(Ab, rA0, kk, laneHi);
      v16bf a1 = ldfragA(Ab, rA1, kk, laneHi);
      acc00 = wmma_bf16(a0, b0, acc00);
      acc01 = wmma_bf16(a0, b1, acc01);
      acc10 = wmma_bf16(a1, b0, acc10);
      acc11 = wmma_bf16(a1, b1, acc11);
    }
  }
  const int gm = gm0 + (w & 1) * 32;
  const int gn = gn0 + (w >> 1) * 32;
  const float bv0 = bias[gn + ln];
  const float bv1 = bias[gn + 16 + ln];
#pragma unroll
  for (int j = 0; j < 8; ++j) {
    int r0 = gm + j + 8 * laneHi;              // C layout: vgpr j = M j / j+8
    C[(size_t)r0 * N + gn + ln]             = acc00[j] + bv0;
    C[(size_t)r0 * N + gn + 16 + ln]        = acc01[j] + bv1;
    C[(size_t)(r0 + 16) * N + gn + ln]      = acc10[j] + bv0;
    C[(size_t)(r0 + 16) * N + gn + 16 + ln] = acc11[j] + bv1;
  }
}

// ---------------------------------------------------------------------------
// Kernel 4: per-batch attention, two-pass online softmax (f32), attn -> bf16
// ---------------------------------------------------------------------------
__global__ __launch_bounds__(512) void k_attn(const float* __restrict__ Qf,
                                              const float* __restrict__ Kf,
                                              const float* __restrict__ Vf,
                                              u16* __restrict__ attnbf) {
  extern __shared__ float smem[];
  float* Qs = smem;                // 16*512
  float* Ks = smem + 8192;
  float* Vs = smem + 16384;
  const int b = blockIdx.x, tid = threadIdx.x;
  const size_t base = (size_t)b * 16 * 512;
#pragma unroll
  for (int i = 0; i < 16; ++i) {
    int idx = i * 512 + tid;
    Qs[idx] = Qf[base + idx];
    Ks[idx] = Kf[base + idx];
    Vs[idx] = Vf[base + idx];
  }
  __syncthreads();
  const int e = tid;
  float q[16];
#pragma unroll
  for (int c = 0; c < 16; ++c) q[c] = Qs[c * 512 + e];
  float m = -3.0e38f, l = 0.f;
  for (int f = 0; f < 512; ++f) {
    float s = 0.f;
#pragma unroll
    for (int c = 0; c < 16; ++c) s += q[c] * Ks[c * 512 + f];
    s *= 0.25f;                                  // 1/sqrt(HID)
    float nm = fmaxf(m, s);
    l = l * __expf(m - nm) + __expf(s - nm);
    m = nm;
  }
  float acc[16];
#pragma unroll
  for (int c = 0; c < 16; ++c) acc[c] = 0.f;
  for (int f = 0; f < 512; ++f) {
    float s = 0.f;
#pragma unroll
    for (int c = 0; c < 16; ++c) s += q[c] * Ks[c * 512 + f];
    float wt = __expf(s * 0.25f - m);
#pragma unroll
    for (int c = 0; c < 16; ++c) acc[c] += wt * Vs[c * 512 + f];
  }
  const float inv = 1.f / l;
#pragma unroll
  for (int c = 0; c < 16; ++c)
    attnbf[base + c * 512 + e] = f2bf(acc[c] * inv);
}

// ---------------------------------------------------------------------------
// Kernel 5: fused out-projection + bo + W3 mixdown + residual.
// Per block: 128 t-rows of one batch. out^T[t,c] = Wo[t,:] . attn[c,:]
// ---------------------------------------------------------------------------
__global__ __launch_bounds__(256) void k_out(const u16* __restrict__ attnbf,
                                             const u16* __restrict__ wobf,
                                             const float* __restrict__ bo,
                                             const float* __restrict__ W3,
                                             const float* __restrict__ x,
                                             float* __restrict__ out) {
  __shared__ __align__(16) u16 sAttn[16 * 520];
  __shared__ __align__(16) u16 sWo[2][128 * 72];
  __shared__ float sOut[128 * 17];
  const int tid = threadIdx.x;
  const int lane = tid & 31, w = tid >> 5;
  const int laneHi = lane >> 4, ln = lane & 15;
  const int b = blockIdx.y, t0 = blockIdx.x * 128;
  const u16* attnb = attnbf + (size_t)b * 16 * 512;

  // attn tile (all K=512) once per block
#pragma unroll
  for (int i = 0; i < 4; ++i) {
    int s = tid + i * 256;
    int r = s >> 6, c16 = s & 63;
    cp16(&sAttn[r * 520 + c16 * 8], attnb + r * 512 + c16 * 8);
  }
  auto issueWo = [&](int buf, int k0) {
#pragma unroll
    for (int i = 0; i < 4; ++i) {
      int s = tid + i * 256;
      int r = s >> 3, sg = s & 7;
      cp16(&sWo[buf][r * 72 + sg * 8],
           wobf + (size_t)(t0 + r) * 512 + k0 + sg * 8);
    }
  };
  issueWo(0, 0);
  v8f acc = {};
  const int rA = w * 16 + ln;
  for (int ko = 0; ko < 8; ++ko) {
    const int buf = ko & 1;
    wait_async();
    __syncthreads();
    if (ko + 1 < 8) issueWo(buf ^ 1, (ko + 1) * 64);
    const u16* Ab = sWo[buf];
#pragma unroll
    for (int kk = 0; kk < 64; kk += 32) {
      v16bf bf = ldfragB(sAttn, ln, 520, ko * 64 + kk, laneHi);
      v16bf a  = ldfragA(Ab, rA, kk, laneHi);
      acc = wmma_bf16(a, bf, acc);
    }
  }
#pragma unroll
  for (int j = 0; j < 8; ++j)
    sOut[(w * 16 + j + 8 * laneHi) * 17 + ln] = acc[j];
  __syncthreads();
  if (tid < 128) {
    int t = t0 + tid;
    float bv = bo[t];
    float o0 = 0.f, o1 = 0.f, o2 = 0.f;
#pragma unroll
    for (int c = 0; c < 16; ++c) {
      float v = sOut[tid * 17 + c] + bv;
      o0 += W3[c] * v;
      o1 += W3[16 + c] * v;
      o2 += W3[32 + c] * v;
    }
    size_t xb = ((size_t)b * 7) * T_ + t;
    size_t ob = ((size_t)b * 3) * T_ + t;
    out[ob]          = x[xb] + o0;
    out[ob + T_]     = x[xb + T_] + o1;
    out[ob + 2 * T_] = x[xb + 2 * T_] + o2;
  }
}

// ---------------------------------------------------------------------------
extern "C" void kernel_launch(void* const* d_in, const int* in_sizes, int n_in,
                              void* d_out, int out_size, void* d_ws, size_t ws_size,
                              hipStream_t stream) {
  (void)in_sizes; (void)n_in; (void)out_size; (void)ws_size;
  const float* x  = (const float*)d_in[0];
  const float* W1 = (const float*)d_in[1];
  const float* W2 = (const float*)d_in[2];
  const float* Wq = (const float*)d_in[3];
  const float* bq = (const float*)d_in[4];
  const float* Wk = (const float*)d_in[5];
  const float* bk = (const float*)d_in[6];
  const float* Wv = (const float*)d_in[7];
  const float* bv = (const float*)d_in[8];
  const float* Wo = (const float*)d_in[9];
  const float* bo = (const float*)d_in[10];
  const float* W3 = (const float*)d_in[11];
  float* out = (float*)d_out;

  char* ws = (char*)d_ws;                       // ~136 MB total
  const size_t nBT16 = (size_t)B_ * 16 * T_;    // 16,777,216
  const size_t nET   = (size_t)E_ * T_;         // 8,388,608
  u16* qbf  = (u16*)ws;  ws += nBT16 * 2;
  u16* ccbf = (u16*)ws;  ws += nBT16 * 2;
  u16* wqbf = (u16*)ws;  ws += nET * 2;
  u16* wkbf = (u16*)ws;  ws += nET * 2;
  u16* wvbf = (u16*)ws;  ws += nET * 2;
  u16* wobf = (u16*)ws;  ws += nET * 2;
  float* Qf = (float*)ws; ws += (size_t)B_ * 16 * E_ * 4;
  float* Kf = (float*)ws; ws += (size_t)B_ * 16 * E_ * 4;
  float* Vf = (float*)ws; ws += (size_t)B_ * 16 * E_ * 4;
  u16* attnbf = (u16*)ws; ws += (size_t)B_ * 16 * E_ * 2;

  k_qcc<<<(B_ * T_) / 256, 256, 0, stream>>>(x, W1, W2, qbf, ccbf);

  int n4 = (int)(nET / 4);
  int cvtBlocks = (n4 + 255) / 256;
  k_cvt4<<<cvtBlocks, 256, 0, stream>>>(Wq, wqbf, n4);
  k_cvt4<<<cvtBlocks, 256, 0, stream>>>(Wk, wkbf, n4);
  k_cvt4<<<cvtBlocks, 256, 0, stream>>>(Wv, wvbf, n4);
  k_cvt4<<<cvtBlocks, 256, 0, stream>>>(Wo, wobf, n4);

  dim3 gg(E_ / 128, (B_ * 16) / 64);            // (4, 16)
  k_gemm_nt<<<gg, 256, 0, stream>>>(qbf,  wqbf, bq, Qf, B_ * 16, E_, T_);
  k_gemm_nt<<<gg, 256, 0, stream>>>(ccbf, wkbf, bk, Kf, B_ * 16, E_, T_);
  k_gemm_nt<<<gg, 256, 0, stream>>>(ccbf, wvbf, bv, Vf, B_ * 16, E_, T_);

  k_attn<<<B_, 512, 3 * 16 * 512 * sizeof(float), stream>>>(Qf, Kf, Vf, attnbf);

  dim3 go(T_ / 128, B_);
  k_out<<<go, 256, 0, stream>>>(attnbf, wobf, bo, W3, x, out);
}
